// MultiHeadAttention_70385924047170
// MI455X (gfx1250) — matrix operand order repaired and checked by reading, not compiled, not run on hardware
//
#include <hip/hip_runtime.h>
#include <hip/hip_bf16.h>
#include <math.h>

// ---------------------------------------------------------------------------
// Types
// ---------------------------------------------------------------------------
typedef __bf16 bf16;
typedef __attribute__((ext_vector_type(8)))  bf16  bf16x8;
typedef __attribute__((ext_vector_type(16))) bf16  bf16x16;
typedef __attribute__((ext_vector_type(8)))  float f32x8;

// ---------------------------------------------------------------------------
// Problem constants (from reference)
// ---------------------------------------------------------------------------
constexpr int   DM    = 1024;          // D_MODEL (also the GEMM K dim)
constexpr int   NH    = 16;
constexpr int   HD    = 64;
constexpr int   BB    = 2;
constexpr int   TT    = 2048;
constexpr int   MROWS = BB * TT;       // 4096
constexpr int   TPAD  = TT + 32;       // padded key length (causal edge tiles)
constexpr float SCALE = 0.125f;        // 1/sqrt(HD)

// ---------------------------------------------------------------------------
// WMMA fragment helpers (CDNA5 16x16x32 bf16 layouts, CDNA5 ISA)
// A (16x32): lane m=lane&15 holds row m; K chunks at (lane>>4)*8 and +16.
// B (32x16) from [n][k] row-major: lane n=lane&15 holds K chunks at
//   (lane>>4)*8 and +16 (same k order as A).
// C/D (16x16 f32): vgpr r, lane l -> element (r + 8*(l>>4), l&15).
// ---------------------------------------------------------------------------
static __device__ inline bf16x16 mkfrag(bf16x8 lo, bf16x8 hi) {
  return __builtin_shufflevector(lo, hi, 0, 1, 2, 3, 4, 5, 6, 7,
                                 8, 9, 10, 11, 12, 13, 14, 15);
}
static __device__ inline bf16x16 load_a_frag(const bf16* base, int ld) {
  const int lane = threadIdx.x & 31;
  const bf16* p = base + (lane & 15) * ld + ((lane >> 4) * 8);
  bf16x8 lo = *(const bf16x8*)(p);
  bf16x8 hi = *(const bf16x8*)(p + 16);
  return mkfrag(lo, hi);
}
static __device__ inline bf16x16 load_b_frag(const bf16* base, int ld) {
  const int lane = threadIdx.x & 31;
  const bf16* p = base + (lane & 15) * ld + ((lane >> 4) * 8);
  bf16x8 lo = *(const bf16x8*)(p);
  bf16x8 hi = *(const bf16x8*)(p + 16);
  return mkfrag(lo, hi);
}
static __device__ inline f32x8 wmma_bf16(bf16x16 a, bf16x16 b, f32x8 c) {
  return __builtin_amdgcn_wmma_f32_16x16x32_bf16(
      /*neg_a=*/false, a, /*neg_b=*/false, b,
      /*c_mod=*/(short)0, c, /*reuse_a=*/false, /*reuse_b=*/false);
}

// ---------------------------------------------------------------------------
// CDNA5 async global->LDS copy (CDNA5 ISA).
// Per-lane: 16B from global addr to LDS addr; same INST_OFFSET applies to both
// sides. LDS address = low 32 bits of the generic shared pointer (aperture
// truncation). Tracked with ASYNCcnt.
// ---------------------------------------------------------------------------
template <int OFF>
static __device__ __forceinline__ void async_ld16(uint32_t lds_off, const bf16* gp) {
  asm volatile("global_load_async_to_lds_b128 %0, %1, off offset:%2"
               :: "v"(lds_off), "v"((unsigned long long)(uintptr_t)gp), "n"(OFF)
               : "memory");
}
static __device__ __forceinline__ void wait_async0() {
  asm volatile("s_wait_asynccnt 0x0" ::: "memory");
}

// ---------------------------------------------------------------------------
// Elementwise cast f32 -> bf16
// ---------------------------------------------------------------------------
__global__ __launch_bounds__(256) void cast_f32_to_bf16(
    const float* __restrict__ s, bf16* __restrict__ d, int n) {
  int i = blockIdx.x * 256 + threadIdx.x;
  if (i < n) d[i] = (bf16)s[i];
}

// ---------------------------------------------------------------------------
// GEMM: C[m][n] = sum_k A[m][k] * W[n][k]    (i.e. X @ W^T, fp32 out)
// A: bf16 [M][DM] row-major; W: bf16 [N][DM] row-major (weights are [o][i]).
// Block tile 128x256, BK=32, 256 threads = 8 waves (2x4), each wave 64x64
// (16 accumulators -> 16 WMMA per 16 fragment ds_loads per K-tile).
// LDS tiles staged with global_load_async_to_lds_b128 (ASYNCcnt).
// ---------------------------------------------------------------------------
constexpr int GBM = 128, GBN = 256, GBK = 32, LPAD = 8, LDK = GBK + LPAD;

__global__ __launch_bounds__(256) void gemm_xwT(
    const bf16* __restrict__ A, const bf16* __restrict__ W,
    float* __restrict__ C, int M, int N) {
  __shared__ bf16 As[GBM][LDK];   // 10240 B
  __shared__ bf16 Bs[GBN][LDK];   // 20480 B
  const int tid  = threadIdx.x;
  const int lane = tid & 31;
  const int wid  = tid >> 5;
  const int m0 = blockIdx.x * GBM, n0 = blockIdx.y * GBN;
  const int wm = (wid >> 2) * 64;   // wave row offset in block tile (0,64)
  const int wn = (wid & 3) * 64;    // wave col offset (0..192)
  const int lrow = tid >> 1;        // 0..127 : A staging row
  const int lcol = (tid & 1) * 16;  // 0 or 16: A staging col (halves)
  const f32x8 z8 = {0.f, 0.f, 0.f, 0.f, 0.f, 0.f, 0.f, 0.f};
  f32x8 acc[4][4];
#pragma unroll
  for (int i = 0; i < 4; ++i)
#pragma unroll
    for (int j = 0; j < 4; ++j) acc[i][j] = z8;

  // async staging addresses: A = half a row per thread, B = one row per thread
  const uint32_t ldsA = (uint32_t)(uintptr_t)&As[lrow][lcol];
  const uint32_t ldsB = (uint32_t)(uintptr_t)&Bs[tid][0];
  const bf16* gA = A + (size_t)(m0 + lrow) * DM + lcol;
  const bf16* gB = W + (size_t)(n0 + tid) * DM;

  for (int k0 = 0; k0 < DM; k0 += GBK) {
    // stage 128x32 A and 256x32 B tiles: 6 x 16B async copies per thread
    async_ld16<0>(ldsA, gA + k0);
    async_ld16<16>(ldsA, gA + k0);
    async_ld16<0>(ldsB, gB + k0);
    async_ld16<16>(ldsB, gB + k0);
    async_ld16<32>(ldsB, gB + k0);
    async_ld16<48>(ldsB, gB + k0);
    wait_async0();
    __syncthreads();

    bf16x16 af[4];
#pragma unroll
    for (int mi = 0; mi < 4; ++mi)
      af[mi] = load_a_frag(&As[wm + mi * 16][0], LDK);
#pragma unroll
    for (int nj = 0; nj < 4; ++nj) {
      bf16x16 bfr = load_b_frag(&Bs[wn + nj * 16][0], LDK);
#pragma unroll
      for (int mi = 0; mi < 4; ++mi)
        acc[mi][nj] = wmma_bf16(af[mi], bfr, acc[mi][nj]);
    }
    __syncthreads();
  }

  const int hi = lane >> 4, nn = lane & 15;
#pragma unroll
  for (int mi = 0; mi < 4; ++mi)
#pragma unroll
    for (int nj = 0; nj < 4; ++nj) {
      const int col = n0 + wn + nj * 16 + nn;
      float* cp = C + (size_t)(m0 + wm + mi * 16 + hi * 8) * N + col;
#pragma unroll
      for (int r = 0; r < 8; ++r) cp[(size_t)r * N] = acc[mi][nj][r];
    }
}

// ---------------------------------------------------------------------------
// Zero the padded key/value tail (t in [TT, TPAD)) so edge tiles are safe.
// 65536 threads.
// ---------------------------------------------------------------------------
__global__ __launch_bounds__(256) void zero_pad_kv(
    bf16* __restrict__ Kb, bf16* __restrict__ Vt) {
  int i = blockIdx.x * 256 + threadIdx.x;   // bh(5) | d(6) | tpad(5)
  int t  = 2048 + (i & 31);
  int d  = (i >> 5) & 63;
  int bh = i >> 11;
  Kb[((size_t)bh * TPAD + t) * HD + d]      = (bf16)0.f;
  Vt[((size_t)bh * HD + d) * TPAD + t]      = (bf16)0.f;
}

// ---------------------------------------------------------------------------
// RoPE (fp32) + repack Q,K:  [bt][1024] fp32 -> bf16 [b,h][t][64]
// One thread per rotation pair. 2^21 threads.
// ---------------------------------------------------------------------------
__global__ __launch_bounds__(256) void rope_rearrange(
    const float* __restrict__ Qlin, const float* __restrict__ Klin,
    const int* __restrict__ pos, bf16* __restrict__ Qb, bf16* __restrict__ Kb) {
  int idx = blockIdx.x * 256 + threadIdx.x;  // b(1)|h(4)|t(11)|pair(5)
  int pr = idx & 31;
  int t  = (idx >> 5) & (TT - 1);
  int h  = (idx >> 16) & (NH - 1);
  int b  = idx >> 20;

  float p   = (float)pos[b * TT + t];
  float inv = __powf(10000.0f, -(float)pr * (1.0f / 32.0f));
  float ang = p * inv;
  float sn, cs;
  __sincosf(ang, &sn, &cs);

  size_t src = ((size_t)(b * TT + t)) * DM + h * HD + 2 * pr;
  float q1 = Qlin[src], q2 = Qlin[src + 1];
  float k1 = Klin[src], k2 = Klin[src + 1];

  size_t dq = (((size_t)(b * NH + h)) * TT + t) * HD + 2 * pr;
  Qb[dq]     = (bf16)(q1 * cs - q2 * sn);
  Qb[dq + 1] = (bf16)(q1 * sn + q2 * cs);
  size_t dk = (((size_t)(b * NH + h)) * TPAD + t) * HD + 2 * pr;
  Kb[dk]     = (bf16)(k1 * cs - k2 * sn);
  Kb[dk + 1] = (bf16)(k1 * sn + k2 * cs);
}

// ---------------------------------------------------------------------------
// V repack (transposed): [bt][1024] fp32 -> bf16 [b,h][d][TPAD]
// ---------------------------------------------------------------------------
__global__ __launch_bounds__(256) void v_transpose(
    const float* __restrict__ Vlin, bf16* __restrict__ Vt) {
  int idx = blockIdx.x * 256 + threadIdx.x;  // b(1)|h(4)|t(11)|d(6)
  int d = idx & 63;
  int t = (idx >> 6) & (TT - 1);
  int h = (idx >> 17) & (NH - 1);
  int b = idx >> 21;
  float v = Vlin[((size_t)(b * TT + t)) * DM + h * HD + d];
  Vt[(((size_t)(b * NH + h)) * HD + d) * TPAD + t] = (bf16)v;
}

// ---------------------------------------------------------------------------
// Flash attention: one wave per (b,h,16-query tile). Online softmax, causal.
// Q,K bf16 [bh][t][64]; V bf16 [bh][64][TPAD]; out bf16 [bt][1024].
// ---------------------------------------------------------------------------
__global__ __launch_bounds__(128) void flash_attn(
    const bf16* __restrict__ Qb, const bf16* __restrict__ Kb,
    const bf16* __restrict__ Vt, bf16* __restrict__ Out) {
  __shared__ bf16 Pl[4][16][40];   // per-wave P tile, padded rows
  const int lane = threadIdx.x & 31;
  const int wid  = threadIdx.x >> 5;
  const int w    = blockIdx.x * 4 + wid;
  const int qt   = w & (TT / 16 - 1);     // 0..127
  const int bh   = w >> 7;                // 0..31
  const int q0   = qt * 16;
  const int hi = lane >> 4, nn = lane & 15;

  const bf16* Q = Qb + (size_t)bh * TT * HD;
  const bf16* K = Kb + (size_t)bh * TPAD * HD;
  const bf16* V = Vt + (size_t)bh * HD * TPAD;

  // Q fragments (K-dim 64 -> two 16x32 fragments), resident in registers
  bf16x16 aq0 = load_a_frag(Q + (size_t)q0 * HD, HD);
  bf16x16 aq1 = load_a_frag(Q + (size_t)q0 * HD + 32, HD);

  const f32x8 z8 = {0.f, 0.f, 0.f, 0.f, 0.f, 0.f, 0.f, 0.f};
  f32x8 o[4];
#pragma unroll
  for (int dj = 0; dj < 4; ++dj) o[dj] = z8;
  float mrow[8], lrow[8];
#pragma unroll
  for (int r = 0; r < 8; ++r) { mrow[r] = -INFINITY; lrow[r] = 0.f; }

  for (int t0 = 0; t0 <= q0 + 15; t0 += 32) {
    // S tile 16(q) x 32(t): two 16x16 WMMA C-tiles, each accumulated over d
    f32x8 sc[2];
#pragma unroll
    for (int j = 0; j < 2; ++j) {
      sc[j] = z8;
      bf16x16 bk0 = load_b_frag(K + (size_t)(t0 + 16 * j) * HD, HD);
      sc[j] = wmma_bf16(aq0, bk0, sc[j]);
      bf16x16 bk1 = load_b_frag(K + (size_t)(t0 + 16 * j) * HD + 32, HD);
      sc[j] = wmma_bf16(aq1, bk1, sc[j]);
    }
    // scale + causal mask + online softmax (rows live in width-16 lane groups)
    float pj[2][8];
#pragma unroll
    for (int r = 0; r < 8; ++r) {
      const int q = q0 + r + 8 * hi;
      float v0 = sc[0][r] * SCALE;
      float v1 = sc[1][r] * SCALE;
      if (t0 + nn > q)      v0 = -INFINITY;
      if (t0 + 16 + nn > q) v1 = -INFINITY;
      float mx = fmaxf(v0, v1);
#pragma unroll
      for (int m = 1; m < 16; m <<= 1) mx = fmaxf(mx, __shfl_xor(mx, m, 32));
      const float mn = fmaxf(mrow[r], mx);
      const float p0 = __expf(v0 - mn);
      const float p1 = __expf(v1 - mn);
      float sum = p0 + p1;
#pragma unroll
      for (int m = 1; m < 16; m <<= 1) sum += __shfl_xor(sum, m, 32);
      const float alpha = __expf(mrow[r] - mn);
      lrow[r] = lrow[r] * alpha + sum;
      mrow[r] = mn;
#pragma unroll
      for (int dj = 0; dj < 4; ++dj) o[dj][r] *= alpha;
      pj[0][r] = p0;
      pj[1][r] = p1;
    }
    // C-layout -> A-fragment layout via per-wave LDS tile (in-order per wave)
#pragma unroll
    for (int j = 0; j < 2; ++j)
#pragma unroll
      for (int r = 0; r < 8; ++r)
        Pl[wid][r + 8 * hi][16 * j + nn] = (bf16)pj[j][r];
    asm volatile("" ::: "memory");
    bf16x16 ap = load_a_frag(&Pl[wid][0][0], 40);
#pragma unroll
    for (int dj = 0; dj < 4; ++dj) {
      bf16x16 bv = load_b_frag(V + (size_t)(dj * 16) * TPAD + t0, TPAD);
      o[dj] = wmma_bf16(ap, bv, o[dj]);
    }
  }

  // epilogue: normalize and write bf16 [bt][h*64+d]
  const int b = bh >> 4, h = bh & 15;
#pragma unroll
  for (int r = 0; r < 8; ++r) {
    const int q = q0 + r + 8 * hi;
    const float inv = 1.0f / lrow[r];
#pragma unroll
    for (int dj = 0; dj < 4; ++dj) {
      const int d = dj * 16 + nn;
      Out[((size_t)(b * TT + q)) * DM + h * HD + d] = (bf16)(o[dj][r] * inv);
    }
  }
}

// ---------------------------------------------------------------------------
// Host-side orchestration
// ---------------------------------------------------------------------------
extern "C" void kernel_launch(void* const* d_in, const int* in_sizes, int n_in,
                              void* d_out, int out_size, void* d_ws, size_t ws_size,
                              hipStream_t stream) {
  (void)in_sizes; (void)n_in; (void)out_size; (void)ws_size;
  const float* x   = (const float*)d_in[0];
  const int*   pos = (const int*)d_in[1];
  const float* Wq  = (const float*)d_in[2];
  const float* Wk  = (const float*)d_in[3];
  const float* Wv  = (const float*)d_in[4];
  const float* Wo  = (const float*)d_in[5];
  float* out = (float*)d_out;

  char* ws = (char*)d_ws;
  size_t off = 0;
  auto take = [&](size_t bytes) -> char* {
    char* p = ws + off;
    off += (bytes + 255) & ~(size_t)255;
    return p;
  };
  bf16*  xb   = (bf16*)take((size_t)MROWS * DM * 2);
  bf16*  wqb  = (bf16*)take((size_t)DM * DM * 2);
  bf16*  wkb  = (bf16*)take((size_t)DM * DM * 2);
  bf16*  wvb  = (bf16*)take((size_t)DM * DM * 2);
  bf16*  wob  = (bf16*)take((size_t)DM * DM * 2);
  float* qlin = (float*)take((size_t)MROWS * DM * 4);
  float* klin = (float*)take((size_t)MROWS * DM * 4);
  float* vlin = (float*)take((size_t)MROWS * DM * 4);
  bf16*  qb   = (bf16*)take((size_t)BB * NH * TT * HD * 2);
  bf16*  kb   = (bf16*)take((size_t)BB * NH * TPAD * HD * 2);
  bf16*  vt   = (bf16*)take((size_t)BB * NH * HD * TPAD * 2);
  bf16*  ao   = (bf16*)take((size_t)MROWS * DM * 2);

  // 1) casts to bf16
  cast_f32_to_bf16<<<(MROWS * DM + 255) / 256, 256, 0, stream>>>(x, xb, MROWS * DM);
  cast_f32_to_bf16<<<(DM * DM + 255) / 256, 256, 0, stream>>>(Wq, wqb, DM * DM);
  cast_f32_to_bf16<<<(DM * DM + 255) / 256, 256, 0, stream>>>(Wk, wkb, DM * DM);
  cast_f32_to_bf16<<<(DM * DM + 255) / 256, 256, 0, stream>>>(Wv, wvb, DM * DM);
  cast_f32_to_bf16<<<(DM * DM + 255) / 256, 256, 0, stream>>>(Wo, wob, DM * DM);

  // 2) QKV projections (WMMA GEMM, fp32 out)
  dim3 ggrid(MROWS / GBM, DM / GBN);
  gemm_xwT<<<ggrid, 256, 0, stream>>>(xb, wqb, qlin, MROWS, DM);
  gemm_xwT<<<ggrid, 256, 0, stream>>>(xb, wkb, klin, MROWS, DM);
  gemm_xwT<<<ggrid, 256, 0, stream>>>(xb, wvb, vlin, MROWS, DM);

  // 3) RoPE + repack (and zero the padded key/value tail)
  zero_pad_kv<<<256, 256, 0, stream>>>(kb, vt);
  rope_rearrange<<<(BB * NH * TT * 32) / 256, 256, 0, stream>>>(qlin, klin, pos, qb, kb);
  v_transpose<<<(BB * NH * TT * HD) / 256, 256, 0, stream>>>(vlin, vt);

  // 4) causal flash attention (WMMA)
  flash_attn<<<(BB * NH * (TT / 16)) / 4, 128, 0, stream>>>(qb, kb, vt, ao);

  // 5) output projection (WMMA GEMM, fp32 out -> d_out)
  gemm_xwT<<<ggrid, 256, 0, stream>>>(ao, wob, out, MROWS, DM);
}
